// RobertaCrf_5068061409877
// MI455X (gfx1250) — compile-verified, matrix operand order checked by reading, not run on hardware
//
#include <hip/hip_runtime.h>
#include <hip/hip_bf16.h>

// ---------------------------------------------------------------------------
// Types for CDNA5 WMMA
// ---------------------------------------------------------------------------
typedef __attribute__((ext_vector_type(16))) __bf16 v16bf;
typedef __attribute__((ext_vector_type(8)))  __bf16 v8bf;
typedef __attribute__((ext_vector_type(4)))  __bf16 v4bf;
typedef __attribute__((ext_vector_type(2)))  __bf16 v2bf;
typedef __attribute__((ext_vector_type(8)))  float  v8f;

#define H_DIM   768
#define S_LEN   512
#define B_SZ    8
#define T_TOK   4096          // B*S
#define NHEAD   12
#define D_HEAD  64
#define FF_DIM  3072
#define NLAYER  12
#define NLAB    3

enum { EPI_NONE = 0, EPI_BIAS = 1, EPI_BIAS_GELU = 2, EPI_SCORES = 3 };

__device__ __forceinline__ float gelu_f(float x) {
    return 0.5f * x * (1.0f + erff(x * 0.7071067811865476f));
}

// Load one 16x32 bf16 fragment row slice from LDS per the CDNA5 ISA layout:
// lanes 0-15 : K = [0..7] then [16..23];  lanes 16-31 : K = [8..15] then [24..31]
__device__ __forceinline__ v16bf frag_ld(const __bf16* rowbase, int kb) {
    v8bf lo = *(const v8bf*)(rowbase + kb);        // ds_load_b128
    v8bf hi = *(const v8bf*)(rowbase + kb + 16);   // ds_load_b128
    return __builtin_shufflevector(lo, hi, 0,1,2,3,4,5,6,7,8,9,10,11,12,13,14,15);
}

__device__ __forceinline__ v4bf pack4(float4 d) {
    v4bf p;
    p[0] = (__bf16)d.x; p[1] = (__bf16)d.y;
    p[2] = (__bf16)d.z; p[3] = (__bf16)d.w;
    return p;                                       // 2x v_cvt_pk_bf16_f32
}

__device__ __forceinline__ v2bf pack2(float a, float b) {
    v2bf p;
    p[0] = (__bf16)a; p[1] = (__bf16)b;
    return p;                                       // 1x v_cvt_pk_bf16_f32
}

__device__ __forceinline__ float block_sum(float v, float* red) {
    int tid = threadIdx.x;
    red[tid] = v; __syncthreads();
    for (int s = 128; s > 0; s >>= 1) {
        if (tid < s) red[tid] += red[tid + s];
        __syncthreads();
    }
    float r = red[0]; __syncthreads();
    return r;
}

// ---------------------------------------------------------------------------
// Generic batched bf16 WMMA GEMM:  C = epi(A[MxK] * B + bias)
//   TRANSB=0: B is [K x N] row-major (weights)   -> staged transposed to LDS
//              (register-transposed 2x4 blocks -> packed ds_store_b32)
//   TRANSB=1: B is [N x K] row-major (A @ B^T)   -> staged directly
// Batch offset for X in {A,B,C}:  (z/nhDiv)*xO + (z%nhDiv)*xI
// Block: 256 threads (8 wave32) as 4(M) x 2(N) waves.
// Block tile: 128(M) x 32*NI(N); wave tile 32 x 16*NI; K-step 32.
// Each wave: 2 x NI v_wmma_f32_16x16x32_bf16 accumulators.
// Pipelined: global loads for step k+1 issued in registers during step k WMMAs.
// ---------------------------------------------------------------------------
template<int EPI, int TRANSB, int NI>
__global__ __launch_bounds__(256) void wmma_gemm(
    const float* __restrict__ Ag, const float* __restrict__ Bg,
    const float* __restrict__ bias, float* __restrict__ Cg,
    int K, int lda, int ldb, int ldc,
    long aO, long aI, long bO, long bI, long cO, long cI, int nhDiv,
    const int* __restrict__ mask, float scale, int Scol)
{
    constexpr int NBLK = 32 * NI;     // block N tile
    constexpr int AIT  = 4;           // 128*32 / 4 floats / 256 threads
    constexpr int BIT  = NBLK / 32;   // NBLK*32 / 4 floats / 256 threads (even)

    const int z  = blockIdx.z;
    const int zo = z / nhDiv, zi = z - zo * nhDiv;
    const float* A = Ag + (size_t)zo * aO + (size_t)zi * aI;
    const float* B = Bg + (size_t)zo * bO + (size_t)zi * bI;
    float*       C = Cg + (size_t)zo * cO + (size_t)zi * cI;

    const int bm   = blockIdx.x * 128;
    const int bn   = blockIdx.y * NBLK;
    const int tid  = threadIdx.x;
    const int lane = tid & 31;
    const int wave = tid >> 5;
    const int wm   = wave >> 1;      // 0..3
    const int wn   = wave & 1;       // 0..1

    __shared__ alignas(16) __bf16 As[128 * 32];
    __shared__ alignas(16) __bf16 Bs[NBLK * 32];

    v8f acc[2][NI] = {};
    float4 ar[AIT], br[BIT];

    auto load_all = [&](int k0) {
        #pragma unroll
        for (int it = 0; it < AIT; ++it) {
            int slot = tid + it * 256;
            int row  = slot >> 3;
            int c4   = (slot & 7) << 2;
            ar[it] = *(const float4*)(A + (size_t)(bm + row) * lda + k0 + c4);
        }
        if (TRANSB) {
            #pragma unroll
            for (int it = 0; it < BIT; ++it) {
                int slot = tid + it * 256;
                int row  = slot >> 3;
                int c4   = (slot & 7) << 2;
                br[it] = *(const float4*)(B + (size_t)(bn + row) * ldb + k0 + c4);
            }
        } else {
            // 2 consecutive K-rows x 4 N-columns per slot (register transpose)
            #pragma unroll
            for (int sp = 0; sp < BIT / 2; ++sp) {
                int slot = tid + sp * 256;
                int k2   = (slot / (NBLK / 4)) << 1;   // 0,2,...,30
                int n4   = (slot % (NBLK / 4)) << 2;
                br[2 * sp + 0] = *(const float4*)(B + (size_t)(k0 + k2 + 0) * ldb + bn + n4);
                br[2 * sp + 1] = *(const float4*)(B + (size_t)(k0 + k2 + 1) * ldb + bn + n4);
            }
        }
    };
    auto store_all = [&]() {
        #pragma unroll
        for (int it = 0; it < AIT; ++it) {
            int slot = tid + it * 256;
            int row  = slot >> 3;
            int c4   = (slot & 7) << 2;
            *(v4bf*)&As[row * 32 + c4] = pack4(ar[it]);   // ds_store_b64
        }
        if (TRANSB) {
            #pragma unroll
            for (int it = 0; it < BIT; ++it) {
                int slot = tid + it * 256;
                int row  = slot >> 3;
                int c4   = (slot & 7) << 2;
                *(v4bf*)&Bs[row * 32 + c4] = pack4(br[it]);
            }
        } else {
            #pragma unroll
            for (int sp = 0; sp < BIT / 2; ++sp) {
                int slot = tid + sp * 256;
                int k2   = (slot / (NBLK / 4)) << 1;
                int n4   = (slot % (NBLK / 4)) << 2;
                const float4 d0 = br[2 * sp + 0];
                const float4 d1 = br[2 * sp + 1];
                // Bs[n][k2..k2+1] = {W[k2][n], W[k2+1][n]} : packed 4B stores
                *(v2bf*)&Bs[(n4 + 0) * 32 + k2] = pack2(d0.x, d1.x);
                *(v2bf*)&Bs[(n4 + 1) * 32 + k2] = pack2(d0.y, d1.y);
                *(v2bf*)&Bs[(n4 + 2) * 32 + k2] = pack2(d0.z, d1.z);
                *(v2bf*)&Bs[(n4 + 3) * 32 + k2] = pack2(d0.w, d1.w);
            }
        }
    };

    const int nK = K >> 5;
    load_all(0);
    for (int kt = 0; kt < nK; ++kt) {
        store_all();
        __syncthreads();
        if (kt + 1 < nK) load_all((kt + 1) << 5);   // overlap with WMMAs below

        const int lr = lane & 15;
        const int kb = (lane < 16) ? 0 : 8;
        v16bf af[2], bf[NI];
        #pragma unroll
        for (int mi = 0; mi < 2; ++mi)
            af[mi] = frag_ld(&As[(wm * 32 + mi * 16 + lr) * 32], kb);
        #pragma unroll
        for (int ni = 0; ni < NI; ++ni)
            bf[ni] = frag_ld(&Bs[(wn * (16 * NI) + ni * 16 + lr) * 32], kb);

        #pragma unroll
        for (int mi = 0; mi < 2; ++mi)
            #pragma unroll
            for (int ni = 0; ni < NI; ++ni)
                acc[mi][ni] = __builtin_amdgcn_wmma_f32_16x16x32_bf16(
                    false, af[mi], false, bf[ni], (short)0, acc[mi][ni], false, false);
        __syncthreads();
    }

    // ---- epilogue: C/D layout: VGPR i -> row (i + (lane>=16)*8), col lane%16 ----
    const int lr     = lane & 15;
    const int rowOff = (lane >> 4) << 3;
    #pragma unroll
    for (int mi = 0; mi < 2; ++mi) {
        #pragma unroll
        for (int ni = 0; ni < NI; ++ni) {
            const int col = bn + wn * (16 * NI) + ni * 16 + lr;
            float badd = 0.0f, maskb = 0.0f;
            if constexpr (EPI == EPI_BIAS || EPI == EPI_BIAS_GELU) badd = bias[col];
            if constexpr (EPI == EPI_SCORES)
                maskb = (1.0f - (float)mask[(z / nhDiv) * Scol + col]) * -1.0e4f;
            #pragma unroll
            for (int i = 0; i < 8; ++i) {
                const int gm = bm + wm * 32 + mi * 16 + rowOff + i;
                float vv = acc[mi][ni][i];
                if constexpr (EPI == EPI_BIAS)           vv += badd;
                else if constexpr (EPI == EPI_BIAS_GELU) vv = gelu_f(vv + badd);
                else if constexpr (EPI == EPI_SCORES)    vv = vv * scale + maskb;
                C[(size_t)gm * ldc + col] = vv;
            }
        }
    }
}

// ---------------------------------------------------------------------------
// Embedding gather + LayerNorm (one block per token, 256 thr, 3 elems/thr)
// ---------------------------------------------------------------------------
__global__ __launch_bounds__(256) void embed_ln(
    const int* __restrict__ ids,
    const float* __restrict__ we, const float* __restrict__ pe,
    const float* __restrict__ te,
    const float* __restrict__ g, const float* __restrict__ b,
    float* __restrict__ h)
{
    const int t = blockIdx.x, tid = threadIdx.x;
    __shared__ float red[256];
    const int id = ids[t];
    const int s  = t & (S_LEN - 1);
    float x[3], lsum = 0.0f, lsq = 0.0f;
    #pragma unroll
    for (int i = 0; i < 3; ++i) {
        const int j = tid + (i << 8);
        float v = we[(size_t)id * H_DIM + j] + pe[(size_t)s * H_DIM + j] + te[j];
        x[i] = v; lsum += v; lsq += v * v;
    }
    const float tot  = block_sum(lsum, red);
    const float tots = block_sum(lsq, red);
    const float mean = tot * (1.0f / H_DIM);
    const float var  = tots * (1.0f / H_DIM) - mean * mean;
    const float inv  = rsqrtf(var + 1e-12f);
    #pragma unroll
    for (int i = 0; i < 3; ++i) {
        const int j = tid + (i << 8);
        h[(size_t)t * H_DIM + j] = (x[i] - mean) * inv * g[j] + b[j];
    }
}

// h = LayerNorm(h + y)  (in-place on h; one block per token)
__global__ __launch_bounds__(256) void add_res_ln(
    float* __restrict__ h, const float* __restrict__ y,
    const float* __restrict__ g, const float* __restrict__ b)
{
    const int t = blockIdx.x, tid = threadIdx.x;
    __shared__ float red[256];
    float x[3], lsum = 0.0f, lsq = 0.0f;
    #pragma unroll
    for (int i = 0; i < 3; ++i) {
        const int j = tid + (i << 8);
        float v = h[(size_t)t * H_DIM + j] + y[(size_t)t * H_DIM + j];
        x[i] = v; lsum += v; lsq += v * v;
    }
    const float tot  = block_sum(lsum, red);
    const float tots = block_sum(lsq, red);
    const float mean = tot * (1.0f / H_DIM);
    const float var  = tots * (1.0f / H_DIM) - mean * mean;
    const float inv  = rsqrtf(var + 1e-12f);
    #pragma unroll
    for (int i = 0; i < 3; ++i) {
        const int j = tid + (i << 8);
        h[(size_t)t * H_DIM + j] = (x[i] - mean) * inv * g[j] + b[j];
    }
}

// Row softmax of width 512 (one block per row; 2 elems/thread)
__global__ __launch_bounds__(256) void softmax512(float* __restrict__ p)
{
    float* x = p + (size_t)blockIdx.x * S_LEN;
    const int tid = threadIdx.x;
    __shared__ float red[256];
    float a0 = x[tid], a1 = x[tid + 256];
    float m = fmaxf(a0, a1);
    red[tid] = m; __syncthreads();
    for (int s = 128; s > 0; s >>= 1) {
        if (tid < s) red[tid] = fmaxf(red[tid], red[tid + s]);
        __syncthreads();
    }
    m = red[0]; __syncthreads();
    float e0 = __expf(a0 - m), e1 = __expf(a1 - m);
    red[tid] = e0 + e1; __syncthreads();
    for (int s = 128; s > 0; s >>= 1) {
        if (tid < s) red[tid] += red[tid + s];
        __syncthreads();
    }
    const float inv = 1.0f / red[0];
    x[tid] = e0 * inv; x[tid + 256] = e1 * inv;
}

// logits[t, 0..2] = h[t] @ Wc + bc   (one block per token)
__global__ __launch_bounds__(256) void classifier_k(
    const float* __restrict__ h, const float* __restrict__ Wc,
    const float* __restrict__ bc, float* __restrict__ logits)
{
    const int t = blockIdx.x, tid = threadIdx.x;
    __shared__ float red[256];
    for (int c = 0; c < NLAB; ++c) {
        float p = 0.0f;
        for (int j = tid; j < H_DIM; j += 256)
            p += h[(size_t)t * H_DIM + j] * Wc[j * NLAB + c];
        const float tot = block_sum(p, red);
        if (tid == 0) logits[(size_t)t * NLAB + c] = tot + bc[c];
    }
}

// CRF Viterbi decode: one block, threads 0..7 each handle one batch row.
__global__ __launch_bounds__(32) void viterbi_k(
    const float* __restrict__ logits, const int* __restrict__ mask,
    const float* __restrict__ startt, const float* __restrict__ endt,
    const float* __restrict__ trans, float* __restrict__ out)
{
    __shared__ unsigned char hist[S_LEN - 1][B_SZ][NLAB];
    const int b = threadIdx.x;
    if (b < B_SZ) {
        float sc[NLAB];
        #pragma unroll
        for (int l = 0; l < NLAB; ++l)
            sc[l] = startt[l] + logits[((size_t)b * S_LEN) * NLAB + l];
        for (int t = 1; t < S_LEN; ++t) {
            float ns[NLAB]; int idx[NLAB];
            #pragma unroll
            for (int j = 0; j < NLAB; ++j) {
                float best = sc[0] + trans[0 * NLAB + j]; int bi = 0;
                #pragma unroll
                for (int i = 1; i < NLAB; ++i) {
                    float v = sc[i] + trans[i * NLAB + j];
                    if (v > best) { best = v; bi = i; }
                }
                ns[j]  = best + logits[((size_t)b * S_LEN + t) * NLAB + j];
                idx[j] = bi;
            }
            const int m = mask[b * S_LEN + t];
            #pragma unroll
            for (int j = 0; j < NLAB; ++j) {
                hist[t - 1][b][j] = (unsigned char)idx[j];
                if (m) sc[j] = ns[j];
            }
        }
        #pragma unroll
        for (int l = 0; l < NLAB; ++l) sc[l] += endt[l];
        int tag = 0; float best = sc[0];
        #pragma unroll
        for (int l = 1; l < NLAB; ++l) if (sc[l] > best) { best = sc[l]; tag = l; }
        out[(size_t)b * S_LEN + (S_LEN - 1)] = (float)tag;
        for (int t = S_LEN - 2; t >= 0; --t) {
            const int prev = hist[t][b][tag];
            if (mask[b * S_LEN + t + 1]) tag = prev;
            out[(size_t)b * S_LEN + t] = (float)tag;
        }
    }
}

// ---------------------------------------------------------------------------
// Host orchestration
// ---------------------------------------------------------------------------
extern "C" void kernel_launch(void* const* d_in, const int* in_sizes, int n_in,
                              void* d_out, int out_size, void* d_ws, size_t ws_size,
                              hipStream_t stream) {
    (void)in_sizes; (void)n_in; (void)out_size; (void)ws_size;

    const int*   ids    = (const int*)d_in[0];
    const int*   amask  = (const int*)d_in[1];
    const float* wemb   = (const float*)d_in[2];
    const float* pemb   = (const float*)d_in[3];
    const float* temb   = (const float*)d_in[4];
    const float* eg     = (const float*)d_in[5];
    const float* ebv    = (const float*)d_in[6];
    const float* Wq     = (const float*)d_in[7];
    const float* bq     = (const float*)d_in[8];
    const float* Wk     = (const float*)d_in[9];
    const float* bk     = (const float*)d_in[10];
    const float* Wv     = (const float*)d_in[11];
    const float* bv     = (const float*)d_in[12];
    const float* Wo     = (const float*)d_in[13];
    const float* bo     = (const float*)d_in[14];
    const float* ln1g   = (const float*)d_in[15];
    const float* ln1b   = (const float*)d_in[16];
    const float* W1     = (const float*)d_in[17];
    const float* b1     = (const float*)d_in[18];
    const float* W2     = (const float*)d_in[19];
    const float* b2     = (const float*)d_in[20];
    const float* ln2g   = (const float*)d_in[21];
    const float* ln2b   = (const float*)d_in[22];
    const float* Wc     = (const float*)d_in[23];
    const float* bcv    = (const float*)d_in[24];
    const float* cstart = (const float*)d_in[25];
    const float* cend   = (const float*)d_in[26];
    const float* ctrans = (const float*)d_in[27];

    float* ws     = (float*)d_ws;
    float* h      = ws;
    float* q      = h    + (size_t)T_TOK * H_DIM;
    float* kbuf   = q    + (size_t)T_TOK * H_DIM;
    float* vbuf   = kbuf + (size_t)T_TOK * H_DIM;
    float* ctx    = vbuf + (size_t)T_TOK * H_DIM;
    float* scores = ctx  + (size_t)T_TOK * H_DIM;     // 96*512*512; reused as FF1
    float* logits = scores + (size_t)B_SZ * NHEAD * S_LEN * S_LEN;

    const long SH = (long)S_LEN * H_DIM;
    const long SS = (long)S_LEN * S_LEN;

    embed_ln<<<T_TOK, 256, 0, stream>>>(ids, wemb, pemb, temb, eg, ebv, h);

    for (int L = 0; L < NLAYER; ++L) {
        const size_t wHH = (size_t)L * H_DIM * H_DIM;
        const size_t wHF = (size_t)L * H_DIM * FF_DIM;
        const size_t oH  = (size_t)L * H_DIM;
        const size_t oF  = (size_t)L * FF_DIM;

        // Q, K, V projections: [4096x768] @ [768x768] + bias
        wmma_gemm<EPI_BIAS, 0, 4><<<dim3(32, 6, 1), 256, 0, stream>>>(
            h, Wq + wHH, bq + oH, q, H_DIM, H_DIM, H_DIM, H_DIM,
            0, 0, 0, 0, 0, 0, 1, nullptr, 0.0f, 0);
        wmma_gemm<EPI_BIAS, 0, 4><<<dim3(32, 6, 1), 256, 0, stream>>>(
            h, Wk + wHH, bk + oH, kbuf, H_DIM, H_DIM, H_DIM, H_DIM,
            0, 0, 0, 0, 0, 0, 1, nullptr, 0.0f, 0);
        wmma_gemm<EPI_BIAS, 0, 4><<<dim3(32, 6, 1), 256, 0, stream>>>(
            h, Wv + wHH, bv + oH, vbuf, H_DIM, H_DIM, H_DIM, H_DIM,
            0, 0, 0, 0, 0, 0, 1, nullptr, 0.0f, 0);

        // scores[b,h] = Q K^T / 8 + maskbias : batched (z = b*12 + head), K=64
        wmma_gemm<EPI_SCORES, 1, 4><<<dim3(4, 4, 96), 256, 0, stream>>>(
            q, kbuf, nullptr, scores, D_HEAD, H_DIM, H_DIM, S_LEN,
            SH, (long)D_HEAD, SH, (long)D_HEAD, 12 * SS, SS, NHEAD,
            amask, 0.125f, S_LEN);

        softmax512<<<B_SZ * NHEAD * S_LEN, 256, 0, stream>>>(scores);

        // ctx[b,h] = attn @ V : M=512, N=64, K=512
        wmma_gemm<EPI_NONE, 0, 2><<<dim3(4, 1, 96), 256, 0, stream>>>(
            scores, vbuf, nullptr, ctx, S_LEN, S_LEN, H_DIM, H_DIM,
            12 * SS, SS, SH, (long)D_HEAD, SH, (long)D_HEAD, NHEAD,
            nullptr, 0.0f, 0);

        // attention output projection -> q (reused), then LN1
        wmma_gemm<EPI_BIAS, 0, 4><<<dim3(32, 6, 1), 256, 0, stream>>>(
            ctx, Wo + wHH, bo + oH, q, H_DIM, H_DIM, H_DIM, H_DIM,
            0, 0, 0, 0, 0, 0, 1, nullptr, 0.0f, 0);
        add_res_ln<<<T_TOK, 256, 0, stream>>>(h, q, ln1g + oH, ln1b + oH);

        // FFN: gelu(h @ W1 + b1) -> scores buf ; @ W2 + b2 -> q ; LN2
        wmma_gemm<EPI_BIAS_GELU, 0, 4><<<dim3(32, 24, 1), 256, 0, stream>>>(
            h, W1 + wHF, b1 + oF, scores, H_DIM, H_DIM, FF_DIM, FF_DIM,
            0, 0, 0, 0, 0, 0, 1, nullptr, 0.0f, 0);
        wmma_gemm<EPI_BIAS, 0, 4><<<dim3(32, 6, 1), 256, 0, stream>>>(
            scores, W2 + (size_t)L * FF_DIM * H_DIM, b2 + oH, q,
            FF_DIM, FF_DIM, H_DIM, H_DIM,
            0, 0, 0, 0, 0, 0, 1, nullptr, 0.0f, 0);
        add_res_ln<<<T_TOK, 256, 0, stream>>>(h, q, ln2g + oH, ln2b + oH);
    }

    classifier_k<<<T_TOK, 256, 0, stream>>>(h, Wc, bcv, logits);
    viterbi_k<<<1, 32, 0, stream>>>(logits, amask, cstart, cend, ctrans,
                                    (float*)d_out);
}
